// LineVectorizer_90263032693077
// MI455X (gfx1250) — matrix operand order repaired and checked by reading, not compiled, not run on hardware
//
#include <hip/hip_runtime.h>
#include <hip/hip_bf16.h>

typedef __attribute__((ext_vector_type(16))) _Float16 v16h;
typedef __attribute__((ext_vector_type(8)))  _Float16 v8h;
typedef __attribute__((ext_vector_type(4)))  _Float16 v4h;
typedef __attribute__((ext_vector_type(8)))  float    v8f;

#define B_      2
#define L_      10000
#define C_IN_   256
#define DLOI    128
#define HW_     65536
#define NPTS0   32
#define DIMFC   1024
#define NROWS   (B_ * L_)          // 20000

// ---------------------------------------------------------------- helpers

// 16-bit A-matrix fragment (16x32, MxK): per ISA layout, lane L holds row
// (L&15); halves 0..7 = K[kb..kb+7], halves 8..15 = K[kb+16..kb+23],
// kb = (L>=16) ? 8 : 0.
__device__ __forceinline__ v16h load_fragA(const _Float16* rowp, int kb) {
    union { v16h v; v8h h[2]; } u;
    u.h[0] = *(const v8h*)(rowp + kb);
    u.h[1] = *(const v8h*)(rowp + kb + 16);
    return u.v;
}

// ---------------------------------------------------------------- kernels

__global__ __launch_bounds__(256)
void cvt_f32_f16(const float* __restrict__ src, _Float16* __restrict__ dst, int n) {
    int i = blockIdx.x * blockDim.x + threadIdx.x;
    if (i < n) dst[i] = (_Float16)src[i];
}

// feature [B][256][65536] f32  ->  featT [(b*65536+p)][256] f16  (LDS tiled)
__global__ __launch_bounds__(256)
void transpose_cvt(const float* __restrict__ f, _Float16* __restrict__ ft) {
    __shared__ _Float16 tile[32][33];
    int b  = blockIdx.z;
    int p0 = blockIdx.x * 32;
    int c0 = blockIdx.y * 32;
    const float* src = f + ((size_t)b * C_IN_ + c0) * HW_ + p0;
#pragma unroll
    for (int i = 0; i < 4; i++) {
        int c = threadIdx.y + i * 8;
        tile[c][threadIdx.x] = (_Float16)src[(size_t)c * HW_ + threadIdx.x];
    }
    __syncthreads();
    _Float16* dst = ft + ((size_t)b * HW_ + p0) * C_IN_ + c0;
#pragma unroll
    for (int i = 0; i < 4; i++) {
        int p = threadIdx.y + i * 8;
        dst[(size_t)p * C_IN_ + threadIdx.x] = tile[threadIdx.x][p];
    }
}

// C[m][n] = act( sum_k A[m][k] * Bw[n][k] + bias[n] ), f16 in / f32 acc / f16 out.
// One wave computes a 32x64 tile (2 M-tiles x 4 N-tiles, 8 WMMAs / K-step).
// Grid must give exactly (M/32)*(N/64) waves (no divergence -> EXEC all ones).
template <int RELU>
__global__ __launch_bounds__(128)
void gemm_wmma(const _Float16* __restrict__ A, const _Float16* __restrict__ Bw,
               const float* __restrict__ bias, _Float16* __restrict__ C,
               int M, int N, int K) {
    const int lane = threadIdx.x & 31;
    const int wid  = (blockIdx.x * blockDim.x + threadIdx.x) >> 5;
    const int ntn  = N >> 6;
    const int mg   = wid / ntn;
    const int ng   = wid - mg * ntn;
    const int m_base = mg * 32;
    const int n_base = ng * 64;
    const int rsel = lane & 15;
    const int hs   = lane >> 4;
    const int kbA  = hs * 8;    // A half-wave K interleave
    const int kbB  = hs * 8;    // B half-wave K interleave (same as A)

    const _Float16* arow0 = A + (size_t)(m_base + rsel) * K;
    const _Float16* arow1 = arow0 + (size_t)16 * K;
    const _Float16* brow[4];
#pragma unroll
    for (int j = 0; j < 4; j++)
        brow[j] = Bw + (size_t)(n_base + j * 16 + rsel) * K + kbB;

    const v8f vzero = {0.f, 0.f, 0.f, 0.f, 0.f, 0.f, 0.f, 0.f};
    v8f acc[2][4];
#pragma unroll
    for (int t = 0; t < 2; t++)
#pragma unroll
        for (int j = 0; j < 4; j++) acc[t][j] = vzero;

    for (int k0 = 0; k0 < K; k0 += 32) {
        v16h a0 = load_fragA(arow0 + k0, kbA);
        v16h a1 = load_fragA(arow1 + k0, kbA);
#pragma unroll
        for (int j = 0; j < 4; j++) {
            v16h bf = load_fragA(brow[j] + k0, 0);
            acc[0][j] = __builtin_amdgcn_wmma_f32_16x16x32_f16(
                false, a0, false, bf, (short)0, acc[0][j], false, false);
            acc[1][j] = __builtin_amdgcn_wmma_f32_16x16x32_f16(
                false, a1, false, bf, (short)0, acc[1][j], false, false);
        }
    }

    // C/D layout: lanes 0-15 -> M = vgpr, lanes 16-31 -> M = 8+vgpr; N = lane&15
#pragma unroll
    for (int j = 0; j < 4; j++) {
        int n = n_base + j * 16 + rsel;
        float bv = bias[n];
#pragma unroll
        for (int t = 0; t < 2; t++) {
            int mb = m_base + t * 16 + hs * 8;
#pragma unroll
            for (int v = 0; v < 8; v++) {
                float val = acc[t][j][v] + bv;
                if (RELU) val = fmaxf(val, 0.0f);
                C[(size_t)(mb + v) * N + n] = (_Float16)val;
            }
        }
    }
}

// one wave per line; lane handles 4 channels; bilinear gather + maxpool(4)
__global__ __launch_bounds__(256)
void sample_pool(const _Float16* __restrict__ x, const float* __restrict__ p,
                 _Float16* __restrict__ feat) {
    int gw   = (blockIdx.x * blockDim.x + threadIdx.x) >> 5;
    int lane = threadIdx.x & 31;
    if (gw >= NROWS) return;
    int b = gw / L_;
    const float* pp = p + (size_t)gw * 4;
    float ax = pp[0], ay = pp[1], bx = pp[2], by = pp[3];
    const _Float16* xb = x + (size_t)b * HW_ * DLOI + lane * 4;

    float mx[4][8];
#pragma unroll
    for (int c = 0; c < 4; c++)
#pragma unroll
        for (int t = 0; t < 8; t++) mx[c][t] = -3.0e38f;

#pragma unroll
    for (int t = 0; t < NPTS0; t++) {
        float lam = (float)t * (1.0f / 31.0f);
        float px = ax * lam + bx * (1.0f - lam) - 0.5f;
        float py = ay * lam + by * (1.0f - lam) - 0.5f;
        float px0 = fminf(fmaxf(floorf(px), 0.0f), 255.0f);
        float py0 = fminf(fmaxf(floorf(py), 0.0f), 255.0f);
        float px1 = fminf(px0 + 1.0f, 255.0f);
        float py1 = fminf(py0 + 1.0f, 255.0f);
        int i0 = (int)px0, j0 = (int)py0, i1 = (int)px1, j1 = (int)py1;
        float w00 = (px1 - px) * (py1 - py);
        float w10 = (px - px0) * (py1 - py);
        float w01 = (px1 - px) * (py - py0);
        float w11 = (px - px0) * (py - py0);
        v4h t00 = *(const v4h*)(xb + (size_t)(i0 * 256 + j0) * DLOI);
        v4h t10 = *(const v4h*)(xb + (size_t)(i1 * 256 + j0) * DLOI);
        v4h t01 = *(const v4h*)(xb + (size_t)(i0 * 256 + j1) * DLOI);
        v4h t11 = *(const v4h*)(xb + (size_t)(i1 * 256 + j1) * DLOI);
        const int tp = t >> 2;
#pragma unroll
        for (int c = 0; c < 4; c++) {
            float v = w00 * (float)t00[c] + w10 * (float)t10[c]
                    + w01 * (float)t01[c] + w11 * (float)t11[c];
            mx[c][tp] = fmaxf(mx[c][tp], v);
        }
    }

    // feat[row][c*8 + t1], row-major K = 1024
    _Float16* out = feat + (size_t)gw * (DLOI * 8) + lane * 32;
#pragma unroll
    for (int c = 0; c < 4; c++) {
        v8h o;
#pragma unroll
        for (int t = 0; t < 8; t++) o[t] = (_Float16)mx[c][t];
        *(v8h*)(out + c * 8) = o;
    }
}

// logits[m][0..2] = h2[m] . w3[j] + b3[j]; one wave per row
__global__ __launch_bounds__(256)
void fc3_kernel(const _Float16* __restrict__ h, const float* __restrict__ w3,
                const float* __restrict__ b3, float* __restrict__ out) {
    int gw   = (blockIdx.x * blockDim.x + threadIdx.x) >> 5;
    int lane = threadIdx.x & 31;
    if (gw >= NROWS) return;
    const _Float16* hr = h + (size_t)gw * DIMFC;
    float s0 = 0.f, s1 = 0.f, s2 = 0.f;
#pragma unroll 4
    for (int i = 0; i < 16; i++) {
        int k = lane * 2 + i * 64;
        float h0 = (float)hr[k], h1v = (float)hr[k + 1];
        s0 += h0 * w3[k]        + h1v * w3[k + 1];
        s1 += h0 * w3[1024 + k] + h1v * w3[1025 + k];
        s2 += h0 * w3[2048 + k] + h1v * w3[2049 + k];
    }
#pragma unroll
    for (int off = 16; off > 0; off >>= 1) {
        s0 += __shfl_down(s0, off, 32);
        s1 += __shfl_down(s1, off, 32);
        s2 += __shfl_down(s2, off, 32);
    }
    if (lane == 0) {
        out[gw * 3 + 0] = s0 + b3[0];
        out[gw * 3 + 1] = s1 + b3[1];
        out[gw * 3 + 2] = s2 + b3[2];
    }
}

// ---------------------------------------------------------------- launch

extern "C" void kernel_launch(void* const* d_in, const int* in_sizes, int n_in,
                              void* d_out, int out_size, void* d_ws, size_t ws_size,
                              hipStream_t stream) {
    const float* feature = (const float*)d_in[0];
    const float* p       = (const float*)d_in[1];
    const float* w_fc1   = (const float*)d_in[2];
    const float* b_fc1   = (const float*)d_in[3];
    const float* w1      = (const float*)d_in[4];
    const float* b1      = (const float*)d_in[5];
    const float* w2      = (const float*)d_in[6];
    const float* b2      = (const float*)d_in[7];
    const float* w3      = (const float*)d_in[8];
    const float* b3      = (const float*)d_in[9];
    float* out = (float*)d_out;

    char* ws = (char*)d_ws;
    _Float16* featT = (_Float16*)(ws);                    // 67,108,864 B (dead after fc1)
    _Float16* xmap  = (_Float16*)(ws + 67108864ull);      // 33,554,432 B
    _Float16* wfc1h = (_Float16*)(ws + 100663296ull);     //     65,536 B
    _Float16* w1h   = (_Float16*)(ws + 100728832ull);     //  2,097,152 B
    _Float16* w2h   = (_Float16*)(ws + 102825984ull);     //  2,097,152 B
    _Float16* feat  = (_Float16*)(ws + 104923136ull);     // 40,960,000 B
    _Float16* h1    = (_Float16*)(ws);                    // reuse featT region
    _Float16* h2    = (_Float16*)(ws + 145883136ull);     // 40,960,000 B

    // weight conversions (f32 -> f16)
    cvt_f32_f16<<<(32768 + 255) / 256, 256, 0, stream>>>(w_fc1, wfc1h, 32768);
    cvt_f32_f16<<<(1048576 + 255) / 256, 256, 0, stream>>>(w1, w1h, 1048576);
    cvt_f32_f16<<<(1048576 + 255) / 256, 256, 0, stream>>>(w2, w2h, 1048576);

    // feature transpose+convert: [B,256,HW] f32 -> [B*HW,256] f16
    transpose_cvt<<<dim3(2048, 8, 2), dim3(32, 8), 0, stream>>>(feature, featT);

    // fc1 GEMM: M=131072, N=128, K=256 -> xmap [B*HW][128] f16
    gemm_wmma<0><<<2048, 128, 0, stream>>>(featT, wfc1h, b_fc1, xmap, 131072, 128, 256);

    // bilinear sample + maxpool -> feat [20000][1024] f16
    sample_pool<<<2500, 256, 0, stream>>>(xmap, p, feat);

    // MLP: two 20000x1024x1024 GEMMs with fused bias+ReLU
    gemm_wmma<1><<<2500, 128, 0, stream>>>(feat, w1h, b1, h1, 20000, 1024, 1024);
    gemm_wmma<1><<<2500, 128, 0, stream>>>(h1, w2h, b2, h2, 20000, 1024, 1024);

    // final 1024 -> 3
    fc3_kernel<<<2500, 256, 0, stream>>>(h2, w3, b3, out);
}
